// MoE_59889023975555
// MI455X (gfx1250) — matrix operand order repaired and checked by reading, not compiled, not run on hardware
//
#include <hip/hip_runtime.h>

#define D_DIM   2048
#define E_NUM   16
#define T_TOK   1024
#define INTER   1408
#define SHINTER 2816
#define LDSW    48          // LDS row stride in halves (96B, 16B-aligned frag reads)

typedef __attribute__((ext_vector_type(16))) __bf16 bf16x16;
typedef __attribute__((ext_vector_type(2)))  __bf16 bf16x2;
typedef __attribute__((ext_vector_type(8)))  float  floatx8;
typedef __attribute__((ext_vector_type(2)))  float  floatx2;

__device__ __forceinline__ unsigned cvt2bf(float x, float y) {
    floatx2 f; f[0] = x; f[1] = y;
    bf16x2 b = __builtin_convertvector(f, bf16x2);   // v_cvt_pk_bf16_f32
    return __builtin_bit_cast(unsigned, b);
}

__device__ __forceinline__ void store4bf(unsigned short* p, float4 f) {
    uint2 v;
    v.x = cvt2bf(f.x, f.y);
    v.y = cvt2bf(f.z, f.w);
    *(uint2*)p = v;                                   // ds_store_b64
}

union FragU { uint4 q[2]; bf16x16 v; };

// A-matrix frag (16x32 bf16): lanes0-15 row m, K{0-7,16-23}; lanes16-31 K{8-15,24-31}
__device__ __forceinline__ bf16x16 ldFragA(const unsigned short* base, int rowbase, int lane) {
    const unsigned short* p = base + (rowbase + (lane & 15)) * LDSW + ((lane >> 4) * 8);
    FragU u;
    u.q[0] = *(const uint4*)p;
    u.q[1] = *(const uint4*)(p + 16);
    return u.v;
}

// B-matrix frag (32x16 bf16, stored as Bᵀ rows = W rows): lanes0-15 K{0-7,16-23}, lanes16-31 K{8-15,24-31}
__device__ __forceinline__ bf16x16 ldFragB(const unsigned short* base, int colbase, int lane) {
    const unsigned short* p = base + (colbase + (lane & 15)) * LDSW + ((lane >> 4) * 8);
    FragU u;
    u.q[0] = *(const uint4*)p;
    u.q[1] = *(const uint4*)(p + 16);
    return u.v;
}

__device__ __forceinline__ floatx8 wmma_bf16(bf16x16 a, bf16x16 b, floatx8 c) {
    return __builtin_amdgcn_wmma_f32_16x16x32_bf16(false, a, false, b, (short)0, c, false, false);
}

__device__ __forceinline__ floatx8 zero8() {
    floatx8 z;
#pragma unroll
    for (int i = 0; i < 8; ++i) z[i] = 0.0f;
    return z;
}

// ---------------- Gate: scores -> softmax -> top2 ----------------
__global__ __launch_bounds__(128) void gate_kernel(const float* __restrict__ X,
                                                   const float* __restrict__ GW,
                                                   int* __restrict__ topi,
                                                   float* __restrict__ topw) {
    const int lane = threadIdx.x & 31;
    const int t = blockIdx.x * 4 + (threadIdx.x >> 5);
    const float* xr = X + (size_t)t * D_DIM;

    float xv[64];
#pragma unroll
    for (int i = 0; i < 64; ++i) xv[i] = xr[i * 32 + lane];

    float sc[E_NUM];
#pragma unroll
    for (int e = 0; e < E_NUM; ++e) {
        const float* g = GW + (size_t)e * D_DIM;
        float s = 0.0f;
#pragma unroll
        for (int i = 0; i < 64; ++i) s += xv[i] * g[i * 32 + lane];
#pragma unroll
        for (int off = 16; off > 0; off >>= 1) s += __shfl_xor(s, off, 32);
        sc[e] = s;
    }
    float m = sc[0];
#pragma unroll
    for (int e = 1; e < E_NUM; ++e) m = fmaxf(m, sc[e]);
    float p[E_NUM], den = 0.0f;
#pragma unroll
    for (int e = 0; e < E_NUM; ++e) { p[e] = __expf(sc[e] - m); den += p[e]; }
    int i0 = 0;
#pragma unroll
    for (int e = 1; e < E_NUM; ++e) if (sc[e] > sc[i0]) i0 = e;
    int i1 = (i0 == 0) ? 1 : 0;
#pragma unroll
    for (int e = 0; e < E_NUM; ++e) if (e != i0 && sc[e] > sc[i1]) i1 = e;
    if (lane == 0) {
        float inv = 1.0f / den;
        topi[2 * t] = i0;     topw[2 * t] = p[i0] * inv;
        topi[2 * t + 1] = i1; topw[2 * t + 1] = p[i1] * inv;
    }
}

// ---------------- Bucketize tokens per expert (64-aligned buckets) ----------------
__global__ __launch_bounds__(1024) void bucket_kernel(const int* __restrict__ topi,
                                                      const float* __restrict__ topw,
                                                      int* __restrict__ slot_token,
                                                      float* __restrict__ slot_w,
                                                      int* __restrict__ cnt,
                                                      int* __restrict__ offs) {
    __shared__ int scnt[E_NUM], soffs[E_NUM], scur[E_NUM];
    const int t = threadIdx.x;
    if (t < E_NUM) { scnt[t] = 0; scur[t] = 0; }
    __syncthreads();
    const int e0 = topi[2 * t], e1 = topi[2 * t + 1];
    atomicAdd(&scnt[e0], 1);
    atomicAdd(&scnt[e1], 1);
    __syncthreads();
    if (t == 0) {
        int run = 0;
        for (int e = 0; e < E_NUM; ++e) { soffs[e] = run; run += (scnt[e] + 63) & ~63; }
    }
    __syncthreads();
    if (t < E_NUM) { cnt[t] = scnt[t]; offs[t] = soffs[t]; }
    int p0 = atomicAdd(&scur[e0], 1);
    slot_token[soffs[e0] + p0] = t; slot_w[soffs[e0] + p0] = topw[2 * t];
    int p1 = atomicAdd(&scur[e1], 1);
    slot_token[soffs[e1] + p1] = t; slot_w[soffs[e1] + p1] = topw[2 * t + 1];
}

// ---------------- Fused up-projection: H = silu(X W1ᵀ + b1) * (X W3ᵀ + b3) ----------------
template <bool ROUTED>
__global__ __launch_bounds__(256) void up_kernel(const float* __restrict__ X,
                                                 const float* __restrict__ W1,
                                                 const float* __restrict__ B1,
                                                 const float* __restrict__ W3,
                                                 const float* __restrict__ B3,
                                                 float* __restrict__ H,
                                                 const int* __restrict__ slot_token,
                                                 const int* __restrict__ cnt,
                                                 const int* __restrict__ offs,
                                                 int Ninter) {
    __shared__ unsigned short sA[2][64 * LDSW];
    __shared__ unsigned short sB1[2][128 * LDSW];
    __shared__ unsigned short sB3[2][128 * LDSW];

    const int e  = ROUTED ? blockIdx.z : 0;
    const int mt = blockIdx.y;
    const int c  = ROUTED ? cnt[e] : T_TOK;
    if (mt * 64 >= c) return;
    const int m0 = (ROUTED ? offs[e] : 0) + mt * 64;
    const int n0 = blockIdx.x * 128;

    const float* w1p = W1 + ((size_t)e * Ninter + n0) * (size_t)D_DIM;
    const float* w3p = W3 + ((size_t)e * Ninter + n0) * (size_t)D_DIM;

    const int tid = threadIdx.x;
    const int lane = tid & 31, wv = tid >> 5;
    const int wm = wv & 1, wn = wv >> 1;
    const int r0 = tid >> 3;            // 0..31
    const int k4 = (tid & 7) * 4;

    int tokRow[2];
#pragma unroll
    for (int it = 0; it < 2; ++it) {
        int r = r0 + it * 32;
        if (ROUTED) tokRow[it] = (mt * 64 + r < c) ? slot_token[m0 + r] : -1;
        else        tokRow[it] = m0 + r;
    }

    float4 ra[2], rb1[4], rb3[4];
    auto loadG = [&](int k0) {
#pragma unroll
        for (int it = 0; it < 2; ++it) {
            int tok = tokRow[it];
            ra[it] = make_float4(0.f, 0.f, 0.f, 0.f);
            if (tok >= 0) ra[it] = *(const float4*)(X + (size_t)tok * D_DIM + k0 + k4);
        }
#pragma unroll
        for (int it = 0; it < 4; ++it) {
            size_t off = (size_t)(r0 + it * 32) * D_DIM + k0 + k4;
            rb1[it] = *(const float4*)(w1p + off);
            rb3[it] = *(const float4*)(w3p + off);
        }
    };
    auto storeL = [&](int b) {
#pragma unroll
        for (int it = 0; it < 2; ++it)
            store4bf(&sA[b][(r0 + it * 32) * LDSW + k4], ra[it]);
#pragma unroll
        for (int it = 0; it < 4; ++it) {
            store4bf(&sB1[b][(r0 + it * 32) * LDSW + k4], rb1[it]);
            store4bf(&sB3[b][(r0 + it * 32) * LDSW + k4], rb3[it]);
        }
    };

    floatx8 acc1[4], acc3[4];
#pragma unroll
    for (int i = 0; i < 4; ++i) { acc1[i] = zero8(); acc3[i] = zero8(); }

    loadG(0);
    storeL(0);
    __syncthreads();

    int buf = 0;
    for (int k0 = 0; k0 < D_DIM; k0 += 32) {
        const bool hasNext = (k0 + 32) < D_DIM;
        if (hasNext) loadG(k0 + 32);

        bf16x16 a0 = ldFragA(sA[buf], wm * 32, lane);
        bf16x16 a1 = ldFragA(sA[buf], wm * 32 + 16, lane);
        bf16x16 p0 = ldFragB(sB1[buf], wn * 32, lane);
        bf16x16 p1 = ldFragB(sB1[buf], wn * 32 + 16, lane);
        bf16x16 q0 = ldFragB(sB3[buf], wn * 32, lane);
        bf16x16 q1 = ldFragB(sB3[buf], wn * 32 + 16, lane);
        acc1[0] = wmma_bf16(a0, p0, acc1[0]);
        acc1[1] = wmma_bf16(a0, p1, acc1[1]);
        acc1[2] = wmma_bf16(a1, p0, acc1[2]);
        acc1[3] = wmma_bf16(a1, p1, acc1[3]);
        acc3[0] = wmma_bf16(a0, q0, acc3[0]);
        acc3[1] = wmma_bf16(a0, q1, acc3[1]);
        acc3[2] = wmma_bf16(a1, q0, acc3[2]);
        acc3[3] = wmma_bf16(a1, q1, acc3[3]);

        if (hasNext) storeL(buf ^ 1);
        __syncthreads();
        buf ^= 1;
    }

#pragma unroll
    for (int tm = 0; tm < 2; ++tm)
#pragma unroll
    for (int tn = 0; tn < 2; ++tn) {
        floatx8 v1 = acc1[tm * 2 + tn];
        floatx8 v3 = acc3[tm * 2 + tn];
        int ncol = n0 + wn * 32 + tn * 16 + (lane & 15);
        float bb1 = B1[(size_t)e * Ninter + ncol];
        float bb3 = B3[(size_t)e * Ninter + ncol];
#pragma unroll
        for (int i = 0; i < 8; ++i) {
            int mloc = wm * 32 + tm * 16 + ((lane >> 4) << 3) + i;
            if (mt * 64 + mloc < c) {
                float t1 = v1[i] + bb1;
                float t3 = v3[i] + bb3;
                float hv = (t1 / (1.0f + __expf(-t1))) * t3;
                H[(size_t)(m0 + mloc) * Ninter + ncol] = hv;
            }
        }
    }
}

// ---------------- Down-projection: OUT (+)= H W2ᵀ + b2  [routed: weighted atomicAdd] ----------------
template <bool ROUTED>
__global__ __launch_bounds__(256) void down_kernel(const float* __restrict__ Hin,
                                                   const float* __restrict__ W2,
                                                   const float* __restrict__ B2,
                                                   float* __restrict__ OUT,
                                                   const int* __restrict__ slot_token,
                                                   const float* __restrict__ slot_w,
                                                   const int* __restrict__ cnt,
                                                   const int* __restrict__ offs,
                                                   int Kin) {
    __shared__ unsigned short sA[2][64 * LDSW];
    __shared__ unsigned short sB[2][128 * LDSW];

    const int e  = ROUTED ? blockIdx.z : 0;
    const int mt = blockIdx.y;
    const int c  = ROUTED ? cnt[e] : T_TOK;
    if (mt * 64 >= c) return;
    const int m0 = (ROUTED ? offs[e] : 0) + mt * 64;
    const int n0 = blockIdx.x * 128;

    const float* w2p = W2 + ((size_t)e * D_DIM + n0) * (size_t)Kin;

    const int tid = threadIdx.x;
    const int lane = tid & 31, wv = tid >> 5;
    const int wm = wv & 1, wn = wv >> 1;
    const int r0 = tid >> 3;
    const int k4 = (tid & 7) * 4;

    float4 ra[2], rb[4];
    auto loadG = [&](int k0) {
#pragma unroll
        for (int it = 0; it < 2; ++it)
            ra[it] = *(const float4*)(Hin + (size_t)(m0 + r0 + it * 32) * Kin + k0 + k4);
#pragma unroll
        for (int it = 0; it < 4; ++it)
            rb[it] = *(const float4*)(w2p + (size_t)(r0 + it * 32) * Kin + k0 + k4);
    };
    auto storeL = [&](int b) {
#pragma unroll
        for (int it = 0; it < 2; ++it)
            store4bf(&sA[b][(r0 + it * 32) * LDSW + k4], ra[it]);
#pragma unroll
        for (int it = 0; it < 4; ++it)
            store4bf(&sB[b][(r0 + it * 32) * LDSW + k4], rb[it]);
    };

    floatx8 acc[4];
#pragma unroll
    for (int i = 0; i < 4; ++i) acc[i] = zero8();

    loadG(0);
    storeL(0);
    __syncthreads();

    int buf = 0;
    for (int k0 = 0; k0 < Kin; k0 += 32) {
        const bool hasNext = (k0 + 32) < Kin;
        if (hasNext) loadG(k0 + 32);

        bf16x16 a0 = ldFragA(sA[buf], wm * 32, lane);
        bf16x16 a1 = ldFragA(sA[buf], wm * 32 + 16, lane);
        bf16x16 b0 = ldFragB(sB[buf], wn * 32, lane);
        bf16x16 b1 = ldFragB(sB[buf], wn * 32 + 16, lane);
        acc[0] = wmma_bf16(a0, b0, acc[0]);
        acc[1] = wmma_bf16(a0, b1, acc[1]);
        acc[2] = wmma_bf16(a1, b0, acc[2]);
        acc[3] = wmma_bf16(a1, b1, acc[3]);

        if (hasNext) storeL(buf ^ 1);
        __syncthreads();
        buf ^= 1;
    }

#pragma unroll
    for (int tm = 0; tm < 2; ++tm)
#pragma unroll
    for (int tn = 0; tn < 2; ++tn) {
        floatx8 v = acc[tm * 2 + tn];
        int ncol = n0 + wn * 32 + tn * 16 + (lane & 15);
        float bb = B2[(size_t)e * D_DIM + ncol];
#pragma unroll
        for (int i = 0; i < 8; ++i) {
            int mloc = wm * 32 + tm * 16 + ((lane >> 4) << 3) + i;
            if (mt * 64 + mloc < c) {
                float val = v[i] + bb;
                if (ROUTED) {
                    int tok = slot_token[m0 + mloc];
                    float w = slot_w[m0 + mloc];
                    atomicAdd(&OUT[(size_t)tok * D_DIM + ncol], val * w);
                } else {
                    OUT[(size_t)(m0 + mloc) * D_DIM + ncol] = val;
                }
            }
        }
    }
}

extern "C" void kernel_launch(void* const* d_in, const int* in_sizes, int n_in,
                              void* d_out, int out_size, void* d_ws, size_t ws_size,
                              hipStream_t stream) {
    (void)in_sizes; (void)n_in; (void)out_size; (void)ws_size;
    const float* x      = (const float*)d_in[0];
    const float* gate_w = (const float*)d_in[1];
    const float* w1     = (const float*)d_in[2];
    const float* b1     = (const float*)d_in[3];
    const float* w2     = (const float*)d_in[4];
    const float* b2     = (const float*)d_in[5];
    const float* w3     = (const float*)d_in[6];
    const float* b3     = (const float*)d_in[7];
    const float* sw1    = (const float*)d_in[8];
    const float* sb1    = (const float*)d_in[9];
    const float* sw2    = (const float*)d_in[10];
    const float* sb2    = (const float*)d_in[11];
    const float* sw3    = (const float*)d_in[12];
    const float* sb3    = (const float*)d_in[13];
    float* out = (float*)d_out;

    char* ws = (char*)d_ws;
    int*   topi       = (int*)ws;                          // 2048 ints
    float* topw       = (float*)(ws + 8192);               // 2048 floats
    int*   cnt        = (int*)(ws + 16384);                // 16
    int*   offs       = (int*)(ws + 16384 + 64);           // 16
    int*   slot_token = (int*)(ws + 16384 + 128);          // 4096
    float* slot_w     = (float*)(ws + 16384 + 128 + 16384);
    float* Hr         = (float*)(ws + 65536);                                  // 4096 x INTER
    float* Hs         = (float*)(ws + 65536 + (size_t)4096 * INTER * 4);       // 1024 x SHINTER

    gate_kernel<<<dim3(T_TOK / 4), dim3(128), 0, stream>>>(x, gate_w, topi, topw);
    bucket_kernel<<<dim3(1), dim3(1024), 0, stream>>>(topi, topw, slot_token, slot_w, cnt, offs);

    // Shared expert (also initializes d_out)
    up_kernel<false><<<dim3(SHINTER / 128, T_TOK / 64, 1), dim3(256), 0, stream>>>(
        x, sw1, sb1, sw3, sb3, Hs, nullptr, nullptr, nullptr, SHINTER);
    down_kernel<false><<<dim3(D_DIM / 128, T_TOK / 64, 1), dim3(256), 0, stream>>>(
        Hs, sw2, sb2, out, nullptr, nullptr, nullptr, nullptr, SHINTER);

    // Routed experts (expert-major grid -> L2 weight reuse), accumulate atomically
    up_kernel<true><<<dim3(INTER / 128, T_TOK / 64, E_NUM), dim3(256), 0, stream>>>(
        x, w1, b1, w3, b3, Hr, slot_token, cnt, offs, INTER);
    down_kernel<true><<<dim3(D_DIM / 128, T_TOK / 64, E_NUM), dim3(256), 0, stream>>>(
        Hr, w2, b2, out, slot_token, slot_w, cnt, offs, INTER);
}
